// SwinTransformerBlock3D_61229053772085
// MI455X (gfx1250) — matrix operand order repaired and checked by reading, not compiled, not run on hardware
//
#include <hip/hip_runtime.h>
#include <hip/hip_bf16.h>

typedef __bf16 bf;
typedef __attribute__((ext_vector_type(16))) __bf16 v16bf;
typedef __attribute__((ext_vector_type(8)))  __bf16 v8bf;
typedef __attribute__((ext_vector_type(8)))  float  v8f;
typedef __attribute__((ext_vector_type(4)))  unsigned int v4u;
typedef __attribute__((ext_vector_type(8)))  int v8i;
typedef __attribute__((ext_vector_type(4)))  int v4i;

#define NTOK   50176          // B*T*H*W
#define CDIM   512
#define NWINT  392            // B * NWIN
#define SCALE_ 0.17677669529663689f

// ---------------------------------------------------------------------------
// shifted-window row -> original token index (inverse of roll + partition)
// ---------------------------------------------------------------------------
__device__ __forceinline__ int win_row_to_token(int row) {
    int w = row >> 7, n = row & 127;
    int b  = w / 196, r = w % 196;
    int tw = r / 49, r2 = r % 49;
    int th = r2 / 7, tww = r2 % 7;
    int wd = n >> 6, wh = (n >> 3) & 7, ww = n & 7;
    int t  = (tw * 2 + wd + 1) & 7;           // + SD mod T
    int hh = th * 8 + wh + 4; if (hh >= 56) hh -= 56;
    int wc = tww * 8 + ww + 4; if (wc >= 56) wc -= 56;
    return ((b * 8 + t) * 56 + hh) * 56 + wc;
}

__device__ __forceinline__ int region_label(int ts, int hs, int ws) {
    int lt = (ts < 6)  ? 0 : ((ts < 7)  ? 1 : 2);
    int lh = (hs < 48) ? 0 : ((hs < 52) ? 1 : 2);
    int lw = (ws < 48) ? 0 : ((ws < 52) ? 1 : 2);
    return lt * 9 + lh * 3 + lw;
}

// load a v16bf A/B fragment as two 16-byte LDS chunks
__device__ __forceinline__ v16bf ldfrag(const bf* p0, const bf* p1) {
    v8bf lo = *(const v8bf*)p0;
    v8bf hi = *(const v8bf*)p1;
    v16bf r;
#pragma unroll
    for (int i = 0; i < 8; ++i) { r[i] = lo[i]; r[i + 8] = hi[i]; }
    return r;
}

// ---------------------------------------------------------------------------
// Tensor Data Mover: DMA a rows x cols bf16 tile (row stride ld elements)
// from global into LDS at byte offset lds_off, inserting 16B of padding per
// 128B row so the LDS image has a 72-element row stride (bank-conflict-free).
// D# layout per CDNA5 ISA ch.8 (group0: count/lds/global/type, group1: dims).
// ---------------------------------------------------------------------------
__device__ __forceinline__ void tdm_tile_load(const bf* gsrc, unsigned lds_off,
                                              int rows, int cols, int ld) {
    unsigned long long ga = (unsigned long long)(uintptr_t)gsrc;
    v4u g0;
    g0.x = 1u;                                   // count=1, user descriptor
    g0.y = lds_off;                              // LDS byte address
    g0.z = (unsigned)ga;                         // global_addr[31:0]
    g0.w = (unsigned)(ga >> 32) | 0x80000000u;   // global_addr[56:32] | type=2
    v8i g1;
    g1[0] = (int)((1u << 16)    // data_size = 2 bytes
                | (1u << 20)    // pad_enable
                | (4u << 22)    // pad_interval: 32 DWORDs (=128B = one row)
                | (3u << 25));  // pad_amount: 4 DWORDs (=16B = 8 bf16)
    g1[1] = (int)(((unsigned)cols & 0xFFFFu) << 16);     // tensor_dim0[15:0]
    g1[2] = (int)(((unsigned)rows & 0xFFFFu) << 16);     // dim0 hi=0 | tensor_dim1[15:0]
    g1[3] = (int)(((unsigned)cols & 0xFFFFu) << 16);     // dim1 hi=0 | tile_dim0
    g1[4] = (int)((unsigned)rows & 0xFFFFu);             // tile_dim1 | tile_dim2=0
    g1[5] = ld;                                          // tensor_dim0_stride[31:0]
    g1[6] = 0;
    g1[7] = 0;
    v4i z4 = {0, 0, 0, 0};
#if defined(__clang_major__) && (__clang_major__ >= 23)
    v8i z8 = {0, 0, 0, 0, 0, 0, 0, 0};
    __builtin_amdgcn_tensor_load_to_lds(g0, g1, z4, z4, z8, 0);
#else
    __builtin_amdgcn_tensor_load_to_lds(g0, g1, z4, z4, 0);
#endif
}

// ---------------------------------------------------------------------------
// fp32 -> bf16 weight conversion
// ---------------------------------------------------------------------------
__global__ void cvt_bf16(const float* __restrict__ s, bf* __restrict__ d, int n) {
    int i = blockIdx.x * blockDim.x + threadIdx.x;
    if (i < n) d[i] = (bf)s[i];
}

// ---------------------------------------------------------------------------
// LayerNorm over C=512.  mode=1: gather through shift+window-partition map.
// ---------------------------------------------------------------------------
__global__ __launch_bounds__(128) void ln_kernel(const float* __restrict__ src,
                                                 const float* __restrict__ g,
                                                 const float* __restrict__ beta,
                                                 bf* __restrict__ dst, int mode) {
    __shared__ float red[8];
    int row = blockIdx.x;
    int tok = mode ? win_row_to_token(row) : row;
    int tid = threadIdx.x;
    float4 v = *((const float4*)(src + (size_t)tok * CDIM) + tid);
    float s  = v.x + v.y + v.z + v.w;
    float s2 = v.x * v.x + v.y * v.y + v.z * v.z + v.w * v.w;
#pragma unroll
    for (int off = 1; off < 32; off <<= 1) {
        s  += __shfl_xor(s,  off, 32);
        s2 += __shfl_xor(s2, off, 32);
    }
    int wv = tid >> 5;
    if ((tid & 31) == 0) { red[wv] = s; red[4 + wv] = s2; }
    __syncthreads();
    float ts  = red[0] + red[1] + red[2] + red[3];
    float ts2 = red[4] + red[5] + red[6] + red[7];
    float mean = ts * (1.0f / CDIM);
    float var  = ts2 * (1.0f / CDIM) - mean * mean;
    float rstd = rsqrtf(var + 1e-5f);
    int c = tid * 4;
    bf* o = dst + (size_t)row * CDIM + c;
    o[0] = (bf)((v.x - mean) * rstd * g[c + 0] + beta[c + 0]);
    o[1] = (bf)((v.y - mean) * rstd * g[c + 1] + beta[c + 1]);
    o[2] = (bf)((v.z - mean) * rstd * g[c + 2] + beta[c + 2]);
    o[3] = (bf)((v.w - mean) * rstd * g[c + 3] + beta[c + 3]);
}

// ---------------------------------------------------------------------------
// bf16 WMMA GEMM:  C[M,N] = A[M,K] * W[N,K]^T + bias   (128x128 tile/block)
// TDM double-buffered LDS staging; 8 waves each own a 32x64 sub-tile.
// epi: 0 = bf16 out, 1 = bf16 GELU out, 2 = fp32 scatter+residual (proj),
//      3 = fp32 residual (fc2)
// ---------------------------------------------------------------------------
__global__ __launch_bounds__(256) void gemm_bf16(const bf* __restrict__ A,
                                                 const bf* __restrict__ Wt,
                                                 const float* __restrict__ bias,
                                                 bf* __restrict__ outb,
                                                 float* __restrict__ outf,
                                                 const float* __restrict__ resid,
                                                 int M, int N, int K, int epi) {
    __shared__ bf As[2][128][72];
    __shared__ bf Bs[2][128][72];
    int tid = threadIdx.x, lane = tid & 31, wv = tid >> 5;
    int hh = lane >> 4, l15 = lane & 15;
    int wm = wv >> 1, wn = wv & 1;
    int m0 = blockIdx.x * 128, n0 = blockIdx.y * 128;

    unsigned aoff = (unsigned)(uintptr_t)&As[0][0][0];
    unsigned boff = (unsigned)(uintptr_t)&Bs[0][0][0];
    const unsigned tileB = 128 * 72 * sizeof(bf);
    const bf* gA = A  + (size_t)m0 * K;
    const bf* gB = Wt + (size_t)n0 * K;

    v8f acc[2][4] = {};
    int nstage = K >> 6;

    // prologue: DMA stage 0 (wave 0 only; TDM ignores EXEC, one issue per wave)
    if (tid < 32) {
        tdm_tile_load(gA, aoff, 128, 64, K);
        tdm_tile_load(gB, boff, 128, 64, K);
    }

    for (int s = 0; s < nstage; ++s) {
        int buf = s & 1;
        if (tid < 32) {
            if (s + 1 < nstage) {
                int nb = buf ^ 1;
                tdm_tile_load(gA + (size_t)(s + 1) * 64, aoff + nb * tileB, 128, 64, K);
                tdm_tile_load(gB + (size_t)(s + 1) * 64, boff + nb * tileB, 128, 64, K);
                __builtin_amdgcn_s_wait_tensorcnt(2);   // stage s landed
            } else {
                __builtin_amdgcn_s_wait_tensorcnt(0);
            }
        }
        __syncthreads();
#pragma unroll
        for (int kk = 0; kk < 64; kk += 32) {
            v16bf a[2], b[4];
            int ca = kk + hh * 8;        // A per-lane split-K layout
            int cb = kk + hh * 8;        // B per-lane split-K layout
#pragma unroll
            for (int i = 0; i < 2; ++i)
                a[i] = ldfrag(&As[buf][wm * 32 + i * 16 + l15][ca],
                              &As[buf][wm * 32 + i * 16 + l15][ca + 16]);
#pragma unroll
            for (int j = 0; j < 4; ++j)
                b[j] = ldfrag(&Bs[buf][wn * 64 + j * 16 + l15][cb],
                              &Bs[buf][wn * 64 + j * 16 + l15][cb + 16]);
#pragma unroll
            for (int i = 0; i < 2; ++i)
#pragma unroll
                for (int j = 0; j < 4; ++j)
                    acc[i][j] = __builtin_amdgcn_wmma_f32_16x16x32_bf16(
                        false, a[i], false, b[j], (short)0, acc[i][j], false, false);
        }
        __syncthreads();   // all waves done with buf before TDM overwrites it
    }

#pragma unroll
    for (int i = 0; i < 2; ++i)
#pragma unroll
        for (int j = 0; j < 4; ++j)
#pragma unroll
            for (int jj = 0; jj < 8; ++jj) {
                int grow = m0 + wm * 32 + i * 16 + jj + hh * 8;
                int gcol = n0 + wn * 64 + j * 16 + l15;
                float v = acc[i][j][jj] + bias[gcol];
                if (epi == 0) {
                    outb[(size_t)grow * N + gcol] = (bf)v;
                } else if (epi == 1) {
                    float ge = 0.5f * v * (1.0f + erff(v * 0.70710678118654752f));
                    outb[(size_t)grow * N + gcol] = (bf)ge;
                } else if (epi == 2) {
                    int tok = win_row_to_token(grow);
                    size_t o = (size_t)tok * N + gcol;
                    outf[o] = resid[o] + v;
                } else {
                    size_t o = (size_t)grow * N + gcol;
                    outf[o] = resid[o] + v;
                }
            }
}

// ---------------------------------------------------------------------------
// Windowed attention: one block per (window, head).  N=128 tokens, DH=32.
// ---------------------------------------------------------------------------
__global__ __launch_bounds__(256) void attn_kernel(const bf* __restrict__ qkv,
                                                   const float* __restrict__ rpb,
                                                   bf* __restrict__ obf) {
    __shared__ bf Qs[128][40];
    __shared__ bf Ks[128][40];
    __shared__ bf Vt[32][136];     // V transposed: [dh][token]
    __shared__ bf Ps[128][136];

    int w = blockIdx.x >> 4, head = blockIdx.x & 15;
    int tid = threadIdx.x, lane = tid & 31, wv = tid >> 5;
    int hh = lane >> 4, l15 = lane & 15;

    int r = w % 196;
    int tw = r / 49, r2 = r % 49;
    int th = r2 / 7, tww = r2 % 7;

    // stage Q/K/V for this (window, head)
    {
        int n = tid >> 1, half = tid & 1;
        const bf* src = qkv + (size_t)(w * 128 + n) * 1536 + head * 32 + half * 16;
        v8bf q0 = *(const v8bf*)(src);        v8bf q1 = *(const v8bf*)(src + 8);
        v8bf k0 = *(const v8bf*)(src + 512);  v8bf k1 = *(const v8bf*)(src + 520);
        v8bf v0 = *(const v8bf*)(src + 1024); v8bf v1 = *(const v8bf*)(src + 1032);
        *(v8bf*)&Qs[n][half * 16]     = q0;  *(v8bf*)&Qs[n][half * 16 + 8] = q1;
        *(v8bf*)&Ks[n][half * 16]     = k0;  *(v8bf*)&Ks[n][half * 16 + 8] = k1;
#pragma unroll
        for (int i = 0; i < 8; ++i) {
            Vt[half * 16 + i][n]     = v0[i];
            Vt[half * 16 + 8 + i][n] = v1[i];
        }
    }
    __syncthreads();

    // S = Q K^T  — each wave owns 16 query rows x all 128 keys
    v16bf aq = ldfrag(&Qs[wv * 16 + l15][hh * 8], &Qs[wv * 16 + l15][16 + hh * 8]);
    v8f s[8];
#pragma unroll
    for (int ct = 0; ct < 8; ++ct) {
        v16bf bk = ldfrag(&Ks[ct * 16 + l15][hh * 8], &Ks[ct * 16 + l15][16 + hh * 8]);
        v8f z = {};
        s[ct] = __builtin_amdgcn_wmma_f32_16x16x32_bf16(
            false, aq, false, bk, (short)0, z, false, false);
    }

    // per-lane column (key) coordinates for each of the 8 tiles
    int cmd[8], cmh[8], cmw[8], cmc[8];
#pragma unroll
    for (int ct = 0; ct < 8; ++ct) {
        int m = ct * 16 + l15;
        cmd[ct] = m >> 6; cmh[ct] = (m >> 3) & 7; cmw[ct] = m & 7;
        cmc[ct] = region_label(tw * 2 + cmd[ct], th * 8 + cmh[ct], tww * 8 + cmw[ct]);
    }

    // scale + rel-pos bias + shift mask + softmax, row by row
#pragma unroll
    for (int j = 0; j < 8; ++j) {
        int n  = wv * 16 + j + hh * 8;           // query token
        int dn = n >> 6, hn = (n >> 3) & 7, wn2 = n & 7;
        int cn = region_label(tw * 2 + dn, th * 8 + hn, tww * 8 + wn2);
        float mx = -1e30f;
#pragma unroll
        for (int ct = 0; ct < 8; ++ct) {
            int idx = (dn - cmd[ct] + 1) * 225 + (hn - cmh[ct] + 7) * 15 + (wn2 - cmw[ct] + 7);
            float v = s[ct][j] * SCALE_ + rpb[idx * 16 + head]
                    + ((cn != cmc[ct]) ? -100.0f : 0.0f);
            s[ct][j] = v;
            mx = fmaxf(mx, v);
        }
#pragma unroll
        for (int off = 1; off < 16; off <<= 1) mx = fmaxf(mx, __shfl_xor(mx, off, 32));
        float sum = 0.0f, e[8];
#pragma unroll
        for (int ct = 0; ct < 8; ++ct) { e[ct] = __expf(s[ct][j] - mx); sum += e[ct]; }
#pragma unroll
        for (int off = 1; off < 16; off <<= 1) sum += __shfl_xor(sum, off, 32);
        float inv = 1.0f / sum;
#pragma unroll
        for (int ct = 0; ct < 8; ++ct)
            Ps[wv * 16 + j + hh * 8][ct * 16 + l15] = (bf)(e[ct] * inv);
    }
    __syncthreads();

    // O = P V  — 16 rows x DH=32, K=128 in 4 steps
    v8f oacc[2] = {};
#pragma unroll
    for (int kk = 0; kk < 128; kk += 32) {
        v16bf ap = ldfrag(&Ps[wv * 16 + l15][kk + hh * 8],
                          &Ps[wv * 16 + l15][kk + 16 + hh * 8]);
#pragma unroll
        for (int nt = 0; nt < 2; ++nt) {
            v16bf bv = ldfrag(&Vt[nt * 16 + l15][kk + hh * 8],
                              &Vt[nt * 16 + l15][kk + 16 + hh * 8]);
            oacc[nt] = __builtin_amdgcn_wmma_f32_16x16x32_bf16(
                false, ap, false, bv, (short)0, oacc[nt], false, false);
        }
    }
#pragma unroll
    for (int nt = 0; nt < 2; ++nt)
#pragma unroll
        for (int j = 0; j < 8; ++j) {
            int rr = wv * 16 + j + hh * 8;
            int cc = head * 32 + nt * 16 + l15;
            obf[(size_t)(w * 128 + rr) * CDIM + cc] = (bf)oacc[nt][j];
        }
}

// ---------------------------------------------------------------------------
extern "C" void kernel_launch(void* const* d_in, const int* in_sizes, int n_in,
                              void* d_out, int out_size, void* d_ws, size_t ws_size,
                              hipStream_t stream) {
    const float* x      = (const float*)d_in[0];
    const float* n1g    = (const float*)d_in[1];
    const float* n1b    = (const float*)d_in[2];
    const float* qkv_w  = (const float*)d_in[3];
    const float* qkv_b  = (const float*)d_in[4];
    const float* rpb    = (const float*)d_in[5];
    const float* proj_w = (const float*)d_in[6];
    const float* proj_b = (const float*)d_in[7];
    const float* n2g    = (const float*)d_in[8];
    const float* n2b    = (const float*)d_in[9];
    const float* fc1_w  = (const float*)d_in[10];
    const float* fc1_b  = (const float*)d_in[11];
    const float* fc2_w  = (const float*)d_in[12];
    const float* fc2_b  = (const float*)d_in[13];

    char* ws = (char*)d_ws;
    size_t offQKV = 0;                                       // 50176*2048 bf16 (qkv, then fc1 out)
    size_t szQKV  = (size_t)NTOK * 2048 * sizeof(bf);
    size_t offH   = offQKV + szQKV;                          // 50176*512 bf16 (ln1, then ln2)
    size_t szH    = (size_t)NTOK * CDIM * sizeof(bf);
    size_t offO   = offH + szH;                              // 50176*512 bf16 (attn out)
    size_t offX   = offO + szH;                              // 50176*512 f32 (x after attn residual)
    size_t szX    = (size_t)NTOK * CDIM * sizeof(float);
    size_t offW   = offX + szX;                              // bf16 weights

    bf*    qkvbuf = (bf*)(ws + offQKV);
    bf*    hbuf   = (bf*)(ws + offH);
    bf*    obuf   = (bf*)(ws + offO);
    float* xres   = (float*)(ws + offX);
    bf*    wqkv   = (bf*)(ws + offW);
    bf*    wproj  = wqkv  + (size_t)1536 * 512;
    bf*    wfc1   = wproj + (size_t)512 * 512;
    bf*    wfc2   = wfc1  + (size_t)2048 * 512;

    cvt_bf16<<<(1536 * 512 + 255) / 256, 256, 0, stream>>>(qkv_w,  wqkv,  1536 * 512);
    cvt_bf16<<<(512  * 512 + 255) / 256, 256, 0, stream>>>(proj_w, wproj, 512 * 512);
    cvt_bf16<<<(2048 * 512 + 255) / 256, 256, 0, stream>>>(fc1_w,  wfc1,  2048 * 512);
    cvt_bf16<<<(512 * 2048 + 255) / 256, 256, 0, stream>>>(fc2_w,  wfc2,  512 * 2048);

    // LN1 fused with cyclic shift + window partition (windowed row order)
    ln_kernel<<<NTOK, 128, 0, stream>>>(x, n1g, n1b, hbuf, 1);

    // QKV projection
    gemm_bf16<<<dim3(NWINT, 12), 256, 0, stream>>>(hbuf, wqkv, qkv_b, qkvbuf,
                                                   nullptr, nullptr, NTOK, 1536, CDIM, 0);

    // windowed multi-head attention (bias + mask + softmax fused)
    attn_kernel<<<NWINT * 16, 256, 0, stream>>>(qkvbuf, rpb, obuf);

    // output projection, fused window-reverse + reverse-shift scatter + residual
    gemm_bf16<<<dim3(NWINT, 4), 256, 0, stream>>>(obuf, wproj, proj_b, nullptr,
                                                  xres, x, NTOK, CDIM, CDIM, 2);

    // LN2
    ln_kernel<<<NTOK, 128, 0, stream>>>(xres, n2g, n2b, hbuf, 0);

    // FC1 + exact GELU
    gemm_bf16<<<dim3(NWINT, 16), 256, 0, stream>>>(hbuf, wfc1, fc1_b, qkvbuf,
                                                   nullptr, nullptr, NTOK, 2048, CDIM, 1);

    // FC2 + residual -> d_out
    gemm_bf16<<<dim3(NWINT, 4), 256, 0, stream>>>(qkvbuf, wfc2, fc2_b, nullptr,
                                                  (float*)d_out, xres, NTOK, CDIM, 2048, 3);
}
